// PointTransformerLayer_420906795555
// MI455X (gfx1250) — compile-verified, matrix operand order checked by reading, not run on hardware
//
#include <hip/hip_runtime.h>
#include <hip/hip_bf16.h>

// ---------------------------------------------------------------------------
// PointTransformerLayer for MI455X (gfx1250), wave32 + WMMA f32_16x16x32_f16
// + async global->LDS staging (global_load_async_to_lds_b128 / ASYNCcnt).
//
// Pipeline (stream-ordered, BN batch-stats force global syncs):
//   k_init       zero stat accumulators
//   k_cvt        f16 copies: XH = x, WH = pre-tiled {wq,wk,wv}, WH1 = ww1,
//                BB = bq|bk|bv concatenated
//   k_qkv        xq/xk/xv = x @ {wq,wk,wv} + b        (WMMA, async staging)
//   k_h          h = (p[idx]-p) @ pw1 + pb1, + stats  (3-channel BN stats)
//   k_rqk_stats  per-channel sum/sumsq of r_qk        (256-channel BN stats)
//   k_w1         w1 = relu(bn1(r_qk)) @ ww1 + wb1     (WMMA) + 32-ch BN stats
//   k_e1         w  = softmax_t(relu(bn2(w1)) @ ww2 + wb2)   (WMMA + shfl)
//   k_e2         out[n,s*32+os] = sum_t w[n,t,os] * (xv[idx]+p_r)[n,t,s*32+os]
//
// p_r (327 MB if materialized) is recomputed everywhere from h (3.8 MB).
// ---------------------------------------------------------------------------

#define NPTS   20000
#define CCH    256
#define NSAMP  16
#define OSCH   32
#define NM     (NPTS * NSAMP)          // 320000 rows for BN stats
#define INV_NM (1.0f / (float)NM)
#define EPS    1e-5f

// ws layout (float offsets)
#define OFF_XQ    0
#define OFF_XK    5120000
#define OFF_XV    10240000
#define OFF_H     15360000              // 960000 floats
#define OFF_W1    16320000              // 10240000 floats
#define OFF_ST    26560000              // 584 floats
#define OFF_XH    26560640              // 5,120,000 f16  (x as f16)
#define OFF_WH    29120640              // 196,608  f16  (wq|wk|wv tiled)
#define OFF_WH1   29218944              // 8,192    f16  (ww1 as [n][k])
#define OFF_BB    29223040              // 768 floats    (bq|bk|bv)
// total floats = 29,223,808  (~116.9 MB)
#define ST_P      0
#define ST_C      8
#define ST_OS     520
#define ST_COUNT  584

typedef __attribute__((ext_vector_type(16))) _Float16 v16h;
typedef __attribute__((ext_vector_type(8)))  float    v8f;

union Frag16 {
    v16h h;
    unsigned int u[8];
    _Float16 e[16];
};

__device__ inline v8f zero8() {
    v8f z;
#pragma unroll
    for (int i = 0; i < 8; ++i) z[i] = 0.0f;
    return z;
}

// K index of the even element of f16 pair v (0..7) for lane-half hi (0/1).
// From ISA 16-bit A 16x32 table: VGPR0..3 = K{0..7}(lanes0-15)/K{8..15}(16-31),
// VGPR4..7 = K{16..23}/K{24..31}.  B assumed mirrored with N in place of M.
__device__ inline int kmap(int v, int hi) {
    return ((v >> 2) << 4) + (hi << 3) + ((v & 3) << 1);
}

// LDS byte offset of a __shared__ pointer (generic addr low 32 bits == LDS off)
__device__ inline unsigned lds_off(const void* p) {
    return (unsigned)(unsigned long long)p;
}

// 16B async copy global->LDS, GVS mode: saddr = SGPR base, voff = byte offset.
__device__ inline void async_copy_b128(unsigned lds_addr, const void* sbase, int voff) {
    asm volatile("global_load_async_to_lds_b128 %0, %1, %2"
                 :: "v"(lds_addr), "v"(voff), "s"(sbase)
                 : "memory");
}
__device__ inline void wait_async0() {
    asm volatile("s_wait_asynccnt 0" ::: "memory");
}

__global__ void k_init(float* st, int n) {
    int i = blockIdx.x * blockDim.x + threadIdx.x;
    if (i < n) st[i] = 0.0f;
}

// ---------------------------------------------------------------------------
// k_cvt: one-time f32 -> f16 conversions into async-friendly layouts.
//   XH[e]                      = x[e]
//   WH[((kc*768+col)*32)+kr]   = wmat[col>>8][(kc*32+kr)*256 + (col&255)]
//   WH1[nc*256+k]              = ww1[k*32+nc]
//   BB[mi*256+c]               = {bq,bk,bv}[mi][c]
// ---------------------------------------------------------------------------
__global__ __launch_bounds__(256) void k_cvt(
    const float* __restrict__ x,
    const float* __restrict__ wq, const float* __restrict__ wk,
    const float* __restrict__ wv, const float* __restrict__ ww1,
    const float* __restrict__ bq, const float* __restrict__ bk,
    const float* __restrict__ bv,
    _Float16* __restrict__ XH, _Float16* __restrict__ WH,
    _Float16* __restrict__ WH1, float* __restrict__ BB)
{
    const int gtid = blockIdx.x * 256 + threadIdx.x;
    const int gsz  = gridDim.x * 256;
    const float* wmat[3] = { wq, wk, wv };
    const float* bmat[3] = { bq, bk, bv };
    for (int e = gtid; e < NPTS * CCH; e += gsz)
        XH[e] = (_Float16)x[e];
    for (int e = gtid; e < 8 * 768 * 32; e += gsz) {
        int kr = e & 31;
        int ck = e >> 5;
        int col = ck % 768;
        int kc  = ck / 768;
        int mi = col >> 8, nc = col & 255;
        WH[e] = (_Float16)wmat[mi][(kc * 32 + kr) * 256 + nc];
    }
    for (int e = gtid; e < 32 * 256; e += gsz) {
        int k = e & 255, nc = e >> 8;
        WH1[e] = (_Float16)ww1[k * 32 + nc];
    }
    for (int e = gtid; e < 768; e += gsz)
        BB[e] = bmat[e >> 8][e & 255];
}

// ---------------------------------------------------------------------------
// k_qkv: one block = one 16-row M tile, 8 waves x 6 output 16x16 tiles
//        (3 matrices x 16 N-tiles = 48 tiles).  K looped in chunks of 32.
//        Both LDS operand tiles staged with global_load_async_to_lds_b128.
// ---------------------------------------------------------------------------
__global__ __launch_bounds__(256) void k_qkv(
    const _Float16* __restrict__ XH, const _Float16* __restrict__ WH,
    const float* __restrict__ BB,
    float* __restrict__ xqkv)      // XQ|XK|XV contiguous, 3 x 5,120,000 floats
{
    __shared__ alignas(16) _Float16 xt[16 * 256];   // A tile, f16, [m][k]
    __shared__ alignas(16) _Float16 wt[768 * 32];   // B chunk,  [col][k]

    const int tid  = threadIdx.x;
    const int lane = tid & 31;
    const int wave = tid >> 5;
    const int m0   = blockIdx.x * 16;

    // async stage A tile: 8 KB contiguous from XH + m0*256 (f16)
    {
        const unsigned xb = lds_off(&xt[0]);
        const _Float16* src = XH + m0 * 256;
#pragma unroll
        for (int i = 0; i < 2; ++i) {
            int off = (tid + i * 256) * 16;
            async_copy_b128(xb + off, src, off);
        }
    }

    v8f acc[6];
#pragma unroll
    for (int i = 0; i < 6; ++i) acc[i] = zero8();

    const int m  = lane & 15;
    const int hi = lane >> 4;
    const unsigned wb = lds_off(&wt[0]);

    for (int kc = 0; kc < 8; ++kc) {
        __syncthreads();                      // previous chunk consumed
        // async stage B chunk: 48 KB contiguous from WH + kc*24576 f16
#pragma unroll
        for (int i = 0; i < 12; ++i) {
            int off = (tid + i * 256) * 16;
            async_copy_b128(wb + off, WH, kc * 49152 + off);
        }
        wait_async0();
        __syncthreads();

        Frag16 a;
#pragma unroll
        for (int v = 0; v < 8; ++v) {
            int k = kmap(v, hi);
            a.u[v] = *(const unsigned int*)&xt[m * 256 + kc * 32 + k];
        }
#pragma unroll
        for (int i = 0; i < 6; ++i) {
            int t = wave * 6 + i;
            int col = ((t >> 4) << 8) + ((t & 15) << 4) + m;  // mat*256 + n
            Frag16 b;
#pragma unroll
            for (int v = 0; v < 8; ++v) {
                int k = kmap(v, hi);
                b.u[v] = *(const unsigned int*)&wt[col * 32 + k];
            }
            acc[i] = __builtin_amdgcn_wmma_f32_16x16x32_f16(
                false, a.h, false, b.h, (short)0, acc[i], false, false);
        }
    }

#pragma unroll
    for (int i = 0; i < 6; ++i) {
        int t   = wave * 6 + i;
        int mi  = t >> 4;
        int nc  = ((t & 15) << 4) + m;
        float bias = BB[(mi << 8) + nc];      // single base: plain global load
#pragma unroll
        for (int r = 0; r < 8; ++r) {
            int mm = m0 + r + hi * 8;
            xqkv[mi * (NPTS * CCH) + mm * 256 + nc] = acc[i][r] + bias;
        }
    }
}

// ---------------------------------------------------------------------------
// k_h: h[n,t,:] = (p[idx]-p) @ pw1 + pb1;  warp-reduced sum/sumsq -> atomics
// ---------------------------------------------------------------------------
__global__ __launch_bounds__(256) void k_h(
    const float* __restrict__ p, const int* __restrict__ idx,
    const float* __restrict__ pw1, const float* __restrict__ pb1,
    float* __restrict__ h, float* __restrict__ st)
{
    int row  = blockIdx.x * 256 + threadIdx.x;   // < 320000
    int lane = threadIdx.x & 31;
    int n = row >> 4;
    int j = idx[row];
    float d0 = p[j * 3 + 0] - p[n * 3 + 0];
    float d1 = p[j * 3 + 1] - p[n * 3 + 1];
    float d2 = p[j * 3 + 2] - p[n * 3 + 2];
    float hv[3];
#pragma unroll
    for (int d = 0; d < 3; ++d) {
        hv[d] = d0 * pw1[0 * 3 + d] + d1 * pw1[1 * 3 + d] + d2 * pw1[2 * 3 + d] + pb1[d];
        h[row * 3 + d] = hv[d];
    }
#pragma unroll
    for (int d = 0; d < 3; ++d) {
        float s = hv[d], q = hv[d] * hv[d];
#pragma unroll
        for (int off = 16; off > 0; off >>= 1) {
            s += __shfl_xor(s, off);
            q += __shfl_xor(q, off);
        }
        if (lane == 0) {
            atomicAdd(&st[ST_P + d], s);
            atomicAdd(&st[ST_P + 3 + d], q);
        }
    }
}

// ---------------------------------------------------------------------------
// k_rqk_stats: one block = 64 rows x 256 channels; per-channel sum/sumsq of
//   r_qk = xk[idx] - xq + p_r  (p_r rebuilt from h + bn_p + pw2)
// ---------------------------------------------------------------------------
__global__ __launch_bounds__(256) void k_rqk_stats(
    const float* __restrict__ xq, const float* __restrict__ xk,
    const int* __restrict__ idx, const float* __restrict__ h,
    const float* __restrict__ pg1, const float* __restrict__ pbe1,
    const float* __restrict__ pw2, const float* __restrict__ pb2,
    float* __restrict__ st)
{
    __shared__ float hbs[64 * 3];
    __shared__ int   jr[64];
    const int row0 = blockIdx.x * 64;
    const int tid  = threadIdx.x;

    if (tid < 64) jr[tid] = idx[row0 + tid];
    if (tid < 192) {
        int rr = tid / 3, d = tid % 3;
        float mu = st[ST_P + d] * INV_NM;
        float var = st[ST_P + 3 + d] * INV_NM - mu * mu;
        float rs = rsqrtf(var + EPS);
        float v = (h[(row0 + rr) * 3 + d] - mu) * rs * pg1[d] + pbe1[d];
        hbs[rr * 3 + d] = fmaxf(v, 0.0f);
    }
    __syncthreads();

    const int c = tid;
    float w0 = pw2[c], w1 = pw2[256 + c], w2 = pw2[512 + c], b = pb2[c];
    float s = 0.0f, q = 0.0f;
    for (int rr = 0; rr < 64; ++rr) {
        int row = row0 + rr;
        int n = row >> 4;
        int j = jr[rr];
        float pr = hbs[rr * 3 + 0] * w0 + hbs[rr * 3 + 1] * w1 + hbs[rr * 3 + 2] * w2 + b;
        float r  = xk[j * 256 + c] - xq[n * 256 + c] + pr;
        s += r; q += r * r;
    }
    atomicAdd(&st[ST_C + c], s);
    atomicAdd(&st[ST_C + 256 + c], q);
}

// ---------------------------------------------------------------------------
// k_w1: one wave = one point n (16 rows = its 16 neighbors).
//   A[t][c] = relu(bn1(r_qk)) built into LDS as f16; ww1 (f16, pre-tiled)
//   staged via async copy; WMMA; accumulates bn2 (OS=32) stats of w1.
// ---------------------------------------------------------------------------
__global__ __launch_bounds__(128) void k_w1(
    const float* __restrict__ xq, const float* __restrict__ xk,
    const int* __restrict__ idx, const float* __restrict__ h,
    const float* __restrict__ pg1, const float* __restrict__ pbe1,
    const float* __restrict__ pw2, const float* __restrict__ pb2,
    const float* __restrict__ wg1, const float* __restrict__ wbe1,
    const _Float16* __restrict__ WH1, const float* __restrict__ wb1,
    float* __restrict__ st, float* __restrict__ w1out)
{
    __shared__ alignas(16) _Float16 As[4 * 16 * 256];  // per-wave A tiles
    __shared__ alignas(16) _Float16 Bs[32 * 256];      // ww1 as [n][k]
    __shared__ float bnsc[256], bnsh[256];

    const int tid  = threadIdx.x;
    const int lane = tid & 31;
    const int wave = tid >> 5;
    const int n    = blockIdx.x * 4 + wave;

    // async stage ww1 f16 (16 KB contiguous)
    {
        const unsigned bb = lds_off(&Bs[0]);
#pragma unroll
        for (int i = 0; i < 8; ++i) {
            int off = (tid + i * 128) * 16;
            async_copy_b128(bb + off, WH1, off);
        }
    }
    for (int c = tid; c < 256; c += 128) {
        float mu = st[ST_C + c] * INV_NM;
        float var = st[ST_C + 256 + c] * INV_NM - mu * mu;
        float rs = rsqrtf(var + EPS);
        float g = wg1[c];
        bnsc[c] = rs * g;
        bnsh[c] = wbe1[c] - mu * rs * g;
    }
    float pmu[3], prs[3];
#pragma unroll
    for (int d = 0; d < 3; ++d) {
        pmu[d] = st[ST_P + d] * INV_NM;
        prs[d] = rsqrtf(st[ST_P + 3 + d] * INV_NM - pmu[d] * pmu[d] + EPS);
    }
    __syncthreads();                  // bnsc/bnsh ready

    _Float16* Aw = &As[wave * 16 * 256];
    for (int t = 0; t < 16; ++t) {
        int row = n * 16 + t;
        int j = idx[row];
        float hb[3];
#pragma unroll
        for (int d = 0; d < 3; ++d)
            hb[d] = fmaxf((h[row * 3 + d] - pmu[d]) * prs[d] * pg1[d] + pbe1[d], 0.0f);
#pragma unroll
        for (int c8 = 0; c8 < 8; ++c8) {
            int c = c8 * 32 + lane;
            float pr = hb[0] * pw2[c] + hb[1] * pw2[256 + c] + hb[2] * pw2[512 + c] + pb2[c];
            float r  = xk[j * 256 + c] - xq[n * 256 + c] + pr;
            r = fmaxf(r * bnsc[c] + bnsh[c], 0.0f);
            Aw[t * 256 + c] = (_Float16)r;
        }
    }
    wait_async0();                    // Bs resident
    __syncthreads();                  // As resident

    const int m  = lane & 15;
    const int hi = lane >> 4;
    v8f c0 = zero8(), c1 = zero8();
    for (int kc = 0; kc < 8; ++kc) {
        Frag16 a, b0, b1;
#pragma unroll
        for (int v = 0; v < 8; ++v) {
            int k = kc * 32 + kmap(v, hi);
            a.u[v]  = *(const unsigned int*)&Aw[m * 256 + k];
            b0.u[v] = *(const unsigned int*)&Bs[m * 256 + k];
            b1.u[v] = *(const unsigned int*)&Bs[(16 + m) * 256 + k];
        }
        c0 = __builtin_amdgcn_wmma_f32_16x16x32_f16(false, a.h, false, b0.h, (short)0, c0, false, false);
        c1 = __builtin_amdgcn_wmma_f32_16x16x32_f16(false, a.h, false, b1.h, (short)0, c1, false, false);
    }

    float bias0 = wb1[m], bias1 = wb1[16 + m];
    float s0 = 0.f, q0 = 0.f, s1 = 0.f, q1 = 0.f;
#pragma unroll
    for (int r = 0; r < 8; ++r) {
        int t = r + hi * 8;
        float v0 = c0[r] + bias0;
        float v1 = c1[r] + bias1;
        w1out[(n * 16 + t) * 32 + m]      = v0;
        w1out[(n * 16 + t) * 32 + 16 + m] = v1;
        s0 += v0; q0 += v0 * v0;
        s1 += v1; q1 += v1 * v1;
    }
    s0 += __shfl_xor(s0, 16); q0 += __shfl_xor(q0, 16);
    s1 += __shfl_xor(s1, 16); q1 += __shfl_xor(q1, 16);
    if (lane < 16) {
        atomicAdd(&st[ST_OS + lane], s0);
        atomicAdd(&st[ST_OS + 32 + lane], q0);
        atomicAdd(&st[ST_OS + 16 + lane], s1);
        atomicAdd(&st[ST_OS + 32 + 16 + lane], q1);
    }
}

// ---------------------------------------------------------------------------
// k_e1: one wave = one point.  w2 = relu(bn2(w1)) @ ww2 + wb2 (M=t=16, K=32),
//       then softmax over t (column-wise: lane-half exchange via shfl_xor 16).
// ---------------------------------------------------------------------------
__global__ __launch_bounds__(256) void k_e1(
    const float* __restrict__ w1, const float* __restrict__ st,
    const float* __restrict__ wg2, const float* __restrict__ wbe2,
    const float* __restrict__ ww2, const float* __restrict__ wb2,
    float* __restrict__ wfin)
{
    const int tid  = threadIdx.x;
    const int lane = tid & 31;
    const int wave = tid >> 5;
    const int n    = blockIdx.x * 8 + wave;
    const int m    = lane & 15;   // t for A, n-col for B
    const int hi   = lane >> 4;

    Frag16 a, b0, b1;
#pragma unroll
    for (int v = 0; v < 8; ++v) {
        int k = kmap(v, hi);
#pragma unroll
        for (int dd = 0; dd < 2; ++dd) {
            int kk = k + dd;
            float mu = st[ST_OS + kk] * INV_NM;
            float var = st[ST_OS + 32 + kk] * INV_NM - mu * mu;
            float rs = rsqrtf(var + EPS);
            float sc = rs * wg2[kk];
            float sh = wbe2[kk] - mu * sc;
            float xv = fmaxf(w1[(n * 16 + m) * 32 + kk] * sc + sh, 0.0f);
            a.e[2 * v + dd]  = (_Float16)xv;
            b0.e[2 * v + dd] = (_Float16)ww2[kk * 32 + m];
            b1.e[2 * v + dd] = (_Float16)ww2[kk * 32 + 16 + m];
        }
    }
    v8f c0 = zero8(), c1 = zero8();
    c0 = __builtin_amdgcn_wmma_f32_16x16x32_f16(false, a.h, false, b0.h, (short)0, c0, false, false);
    c1 = __builtin_amdgcn_wmma_f32_16x16x32_f16(false, a.h, false, b1.h, (short)0, c1, false, false);

    float bias0 = wb2[m], bias1 = wb2[16 + m];
    float v0[8], v1[8];
    float mx0 = -3.4e38f, mx1 = -3.4e38f;
#pragma unroll
    for (int r = 0; r < 8; ++r) {
        v0[r] = c0[r] + bias0; mx0 = fmaxf(mx0, v0[r]);
        v1[r] = c1[r] + bias1; mx1 = fmaxf(mx1, v1[r]);
    }
    mx0 = fmaxf(mx0, __shfl_xor(mx0, 16));
    mx1 = fmaxf(mx1, __shfl_xor(mx1, 16));
    float s0 = 0.f, s1 = 0.f;
#pragma unroll
    for (int r = 0; r < 8; ++r) {
        v0[r] = expf(v0[r] - mx0); s0 += v0[r];
        v1[r] = expf(v1[r] - mx1); s1 += v1[r];
    }
    s0 += __shfl_xor(s0, 16);
    s1 += __shfl_xor(s1, 16);
    float i0 = 1.0f / s0, i1 = 1.0f / s1;
#pragma unroll
    for (int r = 0; r < 8; ++r) {
        int t = r + hi * 8;
        wfin[(n * 16 + t) * 32 + m]      = v0[r] * i0;
        wfin[(n * 16 + t) * 32 + 16 + m] = v1[r] * i1;
    }
}

// ---------------------------------------------------------------------------
// k_e2: out[n, s*32+os] = sum_t wfin[n,t,os] * (xv[idx[n,t]] + p_r)[s*32+os]
// ---------------------------------------------------------------------------
__global__ __launch_bounds__(256) void k_e2(
    const int* __restrict__ idx, const float* __restrict__ xv,
    const float* __restrict__ h, const float* __restrict__ st,
    const float* __restrict__ pg1, const float* __restrict__ pbe1,
    const float* __restrict__ pw2, const float* __restrict__ pb2,
    const float* __restrict__ wfin, float* __restrict__ out)
{
    const int n = blockIdx.x;
    const int c = threadIdx.x;
    const int os = c & 31;

    float pmu[3], prs[3];
#pragma unroll
    for (int d = 0; d < 3; ++d) {
        pmu[d] = st[ST_P + d] * INV_NM;
        prs[d] = rsqrtf(st[ST_P + 3 + d] * INV_NM - pmu[d] * pmu[d] + EPS);
    }
    float w0 = pw2[c], w1c = pw2[256 + c], w2c = pw2[512 + c], bc = pb2[c];

    float acc = 0.0f;
    for (int t = 0; t < 16; ++t) {
        int row = n * 16 + t;
        int j = idx[row];
        float hb0 = fmaxf((h[row * 3 + 0] - pmu[0]) * prs[0] * pg1[0] + pbe1[0], 0.0f);
        float hb1 = fmaxf((h[row * 3 + 1] - pmu[1]) * prs[1] * pg1[1] + pbe1[1], 0.0f);
        float hb2 = fmaxf((h[row * 3 + 2] - pmu[2]) * prs[2] * pg1[2] + pbe1[2], 0.0f);
        float pr = hb0 * w0 + hb1 * w1c + hb2 * w2c + bc;
        float v  = xv[j * 256 + c] + pr;
        acc += wfin[row * 32 + os] * v;
    }
    out[n * 256 + c] = acc;
}

// ---------------------------------------------------------------------------
extern "C" void kernel_launch(void* const* d_in, const int* in_sizes, int n_in,
                              void* d_out, int out_size, void* d_ws, size_t ws_size,
                              hipStream_t stream)
{
    const float* p    = (const float*)d_in[0];
    const float* x    = (const float*)d_in[1];
    const int*   idx  = (const int*)  d_in[2];
    const float* wq   = (const float*)d_in[3];
    const float* bq   = (const float*)d_in[4];
    const float* wk   = (const float*)d_in[5];
    const float* bk   = (const float*)d_in[6];
    const float* wv   = (const float*)d_in[7];
    const float* bv   = (const float*)d_in[8];
    const float* pw1  = (const float*)d_in[9];
    const float* pb1  = (const float*)d_in[10];
    const float* pg1  = (const float*)d_in[11];
    const float* pbe1 = (const float*)d_in[12];
    const float* pw2  = (const float*)d_in[13];
    const float* pb2  = (const float*)d_in[14];
    const float* wg1  = (const float*)d_in[15];
    const float* wbe1 = (const float*)d_in[16];
    const float* ww1  = (const float*)d_in[17];
    const float* wb1  = (const float*)d_in[18];
    const float* wg2  = (const float*)d_in[19];
    const float* wbe2 = (const float*)d_in[20];
    const float* ww2  = (const float*)d_in[21];
    const float* wb2  = (const float*)d_in[22];

    float* ws   = (float*)d_ws;        // needs ~116.9 MB
    float* XQ   = ws + OFF_XQ;
    float* XK   = ws + OFF_XK;
    float* XV   = ws + OFF_XV;
    float* H    = ws + OFF_H;
    float* W1   = ws + OFF_W1;
    float* ST   = ws + OFF_ST;
    _Float16* XH  = (_Float16*)(ws + OFF_XH);
    _Float16* WH  = (_Float16*)(ws + OFF_WH);
    _Float16* WH1 = (_Float16*)(ws + OFF_WH1);
    float* BB   = ws + OFF_BB;
    float* WFIN = ws + OFF_XQ;         // aliases XQ+XK (dead after k_w1)

    k_init<<<3, 256, 0, stream>>>(ST, ST_COUNT);
    k_cvt<<<2048, 256, 0, stream>>>(x, wq, wk, wv, ww1, bq, bk, bv,
                                    XH, WH, WH1, BB);
    k_qkv<<<NPTS / 16, 256, 0, stream>>>(XH, WH, BB, XQ);
    k_h<<<NM / 256, 256, 0, stream>>>(p, idx, pw1, pb1, H, ST);
    k_rqk_stats<<<NM / 64, 256, 0, stream>>>(XQ, XK, idx, H, pg1, pbe1, pw2, pb2, ST);
    k_w1<<<NPTS / 4, 128, 0, stream>>>(XQ, XK, idx, H, pg1, pbe1, pw2, pb2,
                                       wg1, wbe1, WH1, wb1, ST, W1);
    k_e1<<<NPTS / 8, 256, 0, stream>>>(W1, ST, wg2, wbe2, ww2, wb2, WFIN);
    k_e2<<<NPTS, 256, 0, stream>>>(idx, XV, H, ST, pg1, pbe1, pw2, pb2, WFIN,
                                   (float*)d_out);
}
